// AssociationLayer_65609920413966
// MI455X (gfx1250) — matrix operand order repaired and checked by reading, not compiled, not run on hardware
//
#include <hip/hip_runtime.h>
#include <hip/hip_bf16.h>
#include <math.h>

typedef __attribute__((ext_vector_type(16))) __bf16 v16bf;
typedef __attribute__((ext_vector_type(8)))  float  v8f;

// ---------------------------------------------------------------------------
// CDNA5 async global->LDS copy (ASYNCcnt-tracked, no VGPR data path).
// ISA 10.2: generic pointers to LDS carry the LDS byte offset in addr[31:0],
// so the low 32 bits of a __shared__-derived pointer are the VDST operand.
// ---------------------------------------------------------------------------
__device__ __forceinline__ uint32_t lds_byte_off(const void* p) {
  return (uint32_t)(uintptr_t)p;
}
__device__ __forceinline__ void async_copy_b128(const void* lds_dst,
                                                const void* global_src) {
  asm volatile("global_load_async_to_lds_b128 %0, %1, off"
               :
               : "v"(lds_byte_off(lds_dst)),
                 "v"((uint64_t)(uintptr_t)global_src)
               : "memory");
}
__device__ __forceinline__ void wait_async_lds() {
  asm volatile("s_wait_asynccnt 0" ::: "memory");
}

// ---------------------------------------------------------------------------
// WMMA fragment loaders (layouts per CDNA5 ISA 7.12.2, wave32)
// A 16x32 bf16: lane L -> row M=L%16, half=L/16 => two contiguous 16B chunks.
// B 32x16 bf16: lane L -> col N=L%16, half=L/16 => two 16B chunks, same k as A.
// C/D 16x16 f32: lane L -> col N=L%16, VGPR r -> row M=r+8*(L/16).
// ---------------------------------------------------------------------------
__device__ __forceinline__ v16bf load_tile_A(const __bf16* A, int lda,
                                             int mBase, int kBase, int lane) {
  int m    = mBase + (lane & 15);
  int half = lane >> 4;
  const __bf16* p = A + (size_t)m * lda + kBase + 8 * half;
  v16bf a;
  ((uint4*)&a)[0] = *(const uint4*)(p);        // K = kBase+8h .. +7
  ((uint4*)&a)[1] = *(const uint4*)(p + 16);   // K = kBase+16+8h .. +7
  return a;
}

__device__ __forceinline__ v16bf load_tile_B(const __bf16* W, int ldw,
                                             int nBase, int kBase, int lane) {
  int n    = nBase + (lane & 15);
  int half = lane >> 4;
  const __bf16* p = W + (size_t)n * ldw + kBase + 8 * half;
  v16bf b;
  ((uint4*)&b)[0] = *(const uint4*)(p);
  ((uint4*)&b)[1] = *(const uint4*)(p + 16);
  return b;
}

__device__ __forceinline__ v8f wmma_bf16(v16bf a, v16bf b, v8f c) {
  return __builtin_amdgcn_wmma_f32_16x16x32_bf16(false, a, false, b,
                                                 (short)0, c, false, false);
}

__device__ __forceinline__ float wave_sum(float v) {
  #pragma unroll
  for (int off = 16; off > 0; off >>= 1) v += __shfl_xor(v, off, 32);
  return v;
}
__device__ __forceinline__ float wave_max(float v) {
  #pragma unroll
  for (int off = 16; off > 0; off >>= 1) v = fmaxf(v, __shfl_xor(v, off, 32));
  return v;
}

// ---------------------------------------------------------------------------
// f32 -> bf16 cast (weights)
// ---------------------------------------------------------------------------
__global__ void cast_f32_bf16(const float* __restrict__ in,
                              __bf16* __restrict__ out, int n) {
  int i = blockIdx.x * 256 + threadIdx.x;
  if (i < n) out[i] = (__bf16)in[i];
}

// ---------------------------------------------------------------------------
// LayerNorm(C=256) + cast to bf16.  One wave per row, 8 f32/lane.
// ---------------------------------------------------------------------------
__global__ __launch_bounds__(128) void ln_to_bf16(
    const float* __restrict__ x, const float* __restrict__ g,
    const float* __restrict__ bta, __bf16* __restrict__ out, int nrows) {
  int row  = blockIdx.x * 4 + (threadIdx.x >> 5);
  int lane = threadIdx.x & 31;
  if (row >= nrows) return;
  const float* rp = x + (size_t)row * 256;
  float v[8]; float s = 0.f;
  #pragma unroll
  for (int j = 0; j < 8; ++j) { v[j] = rp[j * 32 + lane]; s += v[j]; }
  float mean = wave_sum(s) * (1.0f / 256.0f);
  float vs = 0.f;
  #pragma unroll
  for (int j = 0; j < 8; ++j) { float d = v[j] - mean; vs += d * d; }
  float rstd = rsqrtf(wave_sum(vs) * (1.0f / 256.0f) + 1e-5f);
  #pragma unroll
  for (int j = 0; j < 8; ++j) {
    int c = j * 32 + lane;
    out[(size_t)row * 256 + c] = (__bf16)((v[j] - mean) * rstd * g[c] + bta[c]);
  }
}

// ---------------------------------------------------------------------------
// qkv GEMM: [16384,256] @ [768,256]^T, scatter into q/k [B,H,N,D], v [B,H,D,N] bf16.
// A panel (16x256, 8KB) async-staged to LDS once per block, shared by 4 waves.
// q is pre-scaled by D^-0.5 = 1/8.  grid (M/16, 768/64), block 128 (4 waves).
// ---------------------------------------------------------------------------
__global__ __launch_bounds__(128) void gemm_qkv(
    const __bf16* __restrict__ A, const __bf16* __restrict__ W,
    const float* __restrict__ bias, __bf16* __restrict__ q,
    __bf16* __restrict__ k, __bf16* __restrict__ v) {
  __shared__ __align__(16) __bf16 Ap[16 * 256];
  int wave = threadIdx.x >> 5, lane = threadIdx.x & 31;
  int mTile = blockIdx.x;
  int nTile = blockIdx.y * 4 + wave;

  const char* Ag = (const char*)(A + (size_t)mTile * 16 * 256); // contiguous panel
  #pragma unroll
  for (int i = 0; i < 4; ++i) {                // 8KB = 512 x 16B chunks
    int c = threadIdx.x + i * 128;
    async_copy_b128((const char*)Ap + c * 16, Ag + c * 16);
  }
  wait_async_lds();
  __syncthreads();

  v8f acc = {0.f,0.f,0.f,0.f,0.f,0.f,0.f,0.f};
  #pragma unroll 4
  for (int kk = 0; kk < 256; kk += 32)
    acc = wmma_bf16(load_tile_A(Ap, 256, 0, kk, lane),
                    load_tile_B(W, 256, nTile * 16, kk, lane), acc);
  int o     = nTile * 16 + (lane & 15);
  int which = o >> 8;            // 0:q 1:k 2:v (16 | 64 so uniform per tile)
  int head  = (o >> 6) & 3;
  int dd    = o & 63;
  __bf16* dst = (which == 0) ? q : ((which == 1) ? k : v);
  float scale = (which == 0) ? 0.125f : 1.0f;
  float bv = bias[o];
  int rbase = (lane >> 4) * 8;
  #pragma unroll
  for (int r = 0; r < 8; ++r) {
    int m = mTile * 16 + rbase + r;
    int bb = m >> 10, nn = m & 1023;
    dst[(which == 2) ? (((size_t)(bb * 4 + head)) * 64 + dd) * 1024 + nn : (((size_t)(bb * 4 + head)) * 1024 + nn) * 64 + dd] =
        (__bf16)((acc[r] + bv) * scale);
  }
}

// ---------------------------------------------------------------------------
// Fused attention: per (b,h,16-row tile).  block=128 (4 waves).
// Dyn LDS: S f32 [16][1024] (64KB) + P bf16 [16][1024] (32KB) + invs[16]
//          + Qp bf16 [16][64] (2KB, async-staged q panel).
// S = (q*scale)@k^T ; s = (masked? NEG : s) + K ; P = exp(s-max) bf16 ;
// o = (P @ v) * invsum  (normalization folded into epilogue).
// ---------------------------------------------------------------------------
__global__ __launch_bounds__(128) void attn_kernel(
    const __bf16* __restrict__ q, const __bf16* __restrict__ k,
    const __bf16* __restrict__ v, const float* __restrict__ Kbias,
    const int* __restrict__ n1, const int* __restrict__ n2,
    __bf16* __restrict__ o) {
  extern __shared__ char smem[];
  float*  S    = (float*)smem;                        // 65536 B
  __bf16* P    = (__bf16*)(smem + 65536);             // 32768 B
  float*  invs = (float*)(smem + 65536 + 32768);      // 64 B
  __bf16* Qp   = (__bf16*)(smem + 65536 + 32768 + 64);// 2048 B
  int rowTile = blockIdx.x;      // 0..63
  int h       = blockIdx.y;      // 0..3
  int b       = blockIdx.z;      // 0..15
  int wave = threadIdx.x >> 5, lane = threadIdx.x & 31;
  const __bf16* qh = q + ((size_t)(b * 4 + h)) * 1024 * 64;
  const __bf16* kh = k + ((size_t)(b * 4 + h)) * 1024 * 64;
  const __bf16* vh = v + ((size_t)(b * 4 + h)) * 1024 * 64;

  // async-stage the 16x64 q panel (contiguous 2KB = 128 x 16B chunks)
  {
    const char* qg = (const char*)(qh + (size_t)rowTile * 16 * 64);
    int c = threadIdx.x;   // 128 threads, one chunk each
    async_copy_b128((const char*)Qp + c * 16, qg + c * 16);
    wait_async_lds();
  }
  __syncthreads();

  // ---- scores: 64 key tiles, 16 per wave, K-dim 64 (2 wmma steps) ----
  for (int kt = wave; kt < 64; kt += 4) {
    v8f acc = {0.f,0.f,0.f,0.f,0.f,0.f,0.f,0.f};
    #pragma unroll
    for (int kk = 0; kk < 64; kk += 32)
      acc = wmma_bf16(load_tile_A(Qp, 64, 0, kk, lane),
                      load_tile_B(kh, 64, kt * 16, kk, lane), acc);
    int col = kt * 16 + (lane & 15);
    int rbase = (lane >> 4) * 8;
    #pragma unroll
    for (int r = 0; r < 8; ++r) S[(rbase + r) * 1024 + col] = acc[r];
  }
  __syncthreads();

  // ---- mask + K bias + softmax (4 rows per wave) ----
  int nrc = n1[b] * n2[b];
  for (int rr = 0; rr < 4; ++rr) {
    int r    = wave * 4 + rr;
    int nrow = rowTile * 16 + r;
    const float* Krow = Kbias + ((size_t)b * 1024 + nrow) * 1024;
    bool rowInv = nrow >= nrc;
    float mx = -3.0e38f;
    for (int i = lane; i < 1024; i += 32) {
      float s = S[r * 1024 + i];
      s = ((rowInv || i >= nrc) ? -1.0e10f : s) + Krow[i];
      S[r * 1024 + i] = s;
      mx = fmaxf(mx, s);
    }
    mx = wave_max(mx);
    float sum = 0.f;
    for (int i = lane; i < 1024; i += 32) {
      float p = __expf(S[r * 1024 + i] - mx);
      sum += p;
      P[r * 1024 + i] = (__bf16)p;     // unnormalized probs
    }
    sum = wave_sum(sum);
    if (lane == 0) invs[r] = 1.0f / sum;
  }
  __syncthreads();

  // ---- o = P @ v : one 16-col d-tile per wave, K-dim 1024 ----
  v8f acc = {0.f,0.f,0.f,0.f,0.f,0.f,0.f,0.f};
  for (int kk = 0; kk < 1024; kk += 32)
    acc = wmma_bf16(load_tile_A(P, 1024, 0, kk, lane),
                    load_tile_B(vh, 1024, wave * 16, kk, lane), acc);
  int d = wave * 16 + (lane & 15);
  int rbase = (lane >> 4) * 8;
  #pragma unroll
  for (int r = 0; r < 8; ++r) {
    int lr = rbase + r;
    int nrow = rowTile * 16 + lr;
    o[((size_t)(b * 1024 + nrow)) * 256 + h * 64 + d] = (__bf16)(acc[r] * invs[lr]);
  }
}

// ---------------------------------------------------------------------------
// Generic GEMM  Y = A[M,K]bf16 @ W[Nout,K]^T bf16 + bias, with optional
// exact GELU, optional f32 residual add, f32 and/or bf16 outputs.
// A panel (16xK, up to 32KB) async-staged to LDS once per block.
// grid (M/16, Nout/64), block 128 (one 16x16 tile per wave).
// ---------------------------------------------------------------------------
__global__ __launch_bounds__(128) void gemm_bias_act(
    const __bf16* __restrict__ A, const __bf16* __restrict__ W,
    const float* __restrict__ bias, const float* __restrict__ residual,
    float* __restrict__ outF, __bf16* __restrict__ outB,
    int K, int Nout, int gelu) {
  __shared__ __align__(16) __bf16 Ap[16 * 1024];   // 32KB max (K<=1024)
  int wave = threadIdx.x >> 5, lane = threadIdx.x & 31;
  int mTile = blockIdx.x;
  int nTile = blockIdx.y * 4 + wave;

  const char* Ag = (const char*)(A + (size_t)mTile * 16 * K); // contiguous panel
  for (int c = threadIdx.x; c < 2 * K; c += 128)   // 32*K bytes = 2K x 16B
    async_copy_b128((const char*)Ap + c * 16, Ag + c * 16);
  wait_async_lds();
  __syncthreads();

  v8f acc = {0.f,0.f,0.f,0.f,0.f,0.f,0.f,0.f};
  for (int kk = 0; kk < K; kk += 32)
    acc = wmma_bf16(load_tile_A(Ap, K, 0, kk, lane),
                    load_tile_B(W, K, nTile * 16, kk, lane), acc);
  int n = nTile * 16 + (lane & 15);
  int rbase = (lane >> 4) * 8;
  float bv = bias[n];
  #pragma unroll
  for (int r = 0; r < 8; ++r) {
    size_t m = (size_t)mTile * 16 + rbase + r;
    float y = acc[r] + bv;
    if (gelu) y = 0.5f * y * (1.0f + erff(y * 0.70710678118654752f));
    if (residual) y += residual[m * (size_t)Nout + n];
    if (outF) outF[m * (size_t)Nout + n] = y;
    if (outB) outB[m * (size_t)Nout + n] = (__bf16)y;
  }
}

// ---------------------------------------------------------------------------
extern "C" void kernel_launch(void* const* d_in, const int* in_sizes, int n_in,
                              void* d_out, int out_size, void* d_ws, size_t ws_size,
                              hipStream_t stream) {
  (void)in_sizes; (void)n_in; (void)out_size; (void)ws_size;
  const int Bb = 16, Nn = 1024;
  const size_t M = (size_t)Bb * Nn;   // 16384 rows

  const float* x      = (const float*)d_in[0];
  const float* Kbias  = (const float*)d_in[1];
  const int*   n1     = (const int*)d_in[2];
  const int*   n2     = (const int*)d_in[3];
  const float* qkv_w  = (const float*)d_in[4];
  const float* qkv_b  = (const float*)d_in[5];
  const float* proj_w = (const float*)d_in[6];
  const float* proj_b = (const float*)d_in[7];
  const float* ln1_g  = (const float*)d_in[8];
  const float* ln1_b  = (const float*)d_in[9];
  const float* ln2_g  = (const float*)d_in[10];
  const float* ln2_b  = (const float*)d_in[11];
  const float* fc1_w  = (const float*)d_in[12];
  const float* fc1_b  = (const float*)d_in[13];
  const float* fc2_w  = (const float*)d_in[14];
  const float* fc2_b  = (const float*)d_in[15];

  char* ws = (char*)d_ws;
  size_t off = 0;
  auto take = [&](size_t bytes) -> char* {
    char* p = ws + off;
    off += (bytes + 255) & ~(size_t)255;
    return p;
  };
  __bf16* wqkv = (__bf16*)take((size_t)768 * 256 * 2);
  __bf16* wprj = (__bf16*)take((size_t)256 * 256 * 2);
  __bf16* wfc1 = (__bf16*)take((size_t)1024 * 256 * 2);
  __bf16* wfc2 = (__bf16*)take((size_t)256 * 1024 * 2);
  __bf16* hbf  = (__bf16*)take(M * 256 * 2);      // LN1 out, reused for LN2 out
  __bf16* qb   = (__bf16*)take(M * 256 * 2);      // [B,H,N,D]
  __bf16* kb   = (__bf16*)take(M * 256 * 2);
  __bf16* vb   = (__bf16*)take(M * 256 * 2);
  __bf16* obf  = (__bf16*)take(M * 256 * 2);      // attention out [B,N,C]
  float*  X2   = (float*) take(M * 256 * 4);      // x + proj(o)
  __bf16* mbf  = (__bf16*)take(M * 1024 * 2);     // gelu(fc1) bf16

  // weight casts
  cast_f32_bf16<<<(768 * 256 + 255) / 256, 256, 0, stream>>>(qkv_w, wqkv, 768 * 256);
  cast_f32_bf16<<<(256 * 256 + 255) / 256, 256, 0, stream>>>(proj_w, wprj, 256 * 256);
  cast_f32_bf16<<<(1024 * 256 + 255) / 256, 256, 0, stream>>>(fc1_w, wfc1, 1024 * 256);
  cast_f32_bf16<<<(256 * 1024 + 255) / 256, 256, 0, stream>>>(fc2_w, wfc2, 256 * 1024);

  // LN1 -> bf16
  ln_to_bf16<<<(int)(M / 4), 128, 0, stream>>>(x, ln1_g, ln1_b, hbf, (int)M);

  // qkv projection (q pre-scaled by 1/sqrt(D))
  gemm_qkv<<<dim3(1024, 12), 128, 0, stream>>>(hbf, wqkv, qkv_b, qb, kb, vb);

  // fused attention; ~98KB dynamic LDS (CDNA5 WGP has 320KB)
  const int ATTN_SMEM = 65536 + 32768 + 64 + 2048;
  (void)hipFuncSetAttribute((const void*)attn_kernel,
                            hipFuncAttributeMaxDynamicSharedMemorySize, ATTN_SMEM);
  attn_kernel<<<dim3(64, 4, 16), 128, ATTN_SMEM, stream>>>(qb, kb, vb, Kbias,
                                                           n1, n2, obf);

  // proj + residual -> X2 (f32)
  gemm_bias_act<<<dim3(1024, 4), 128, 0, stream>>>(obf, wprj, proj_b, x,
                                                   X2, nullptr, 256, 256, 0);
  // LN2 -> bf16 (reuse hbf)
  ln_to_bf16<<<(int)(M / 4), 128, 0, stream>>>(X2, ln2_g, ln2_b, hbf, (int)M);

  // fc1 + exact GELU -> bf16
  gemm_bias_act<<<dim3(1024, 16), 128, 0, stream>>>(hbf, wfc1, fc1_b, nullptr,
                                                    nullptr, mbf, 256, 1024, 1);
  // fc2 + residual -> d_out (f32)
  gemm_bias_act<<<dim3(1024, 4), 128, 0, stream>>>(mbf, wfc2, fc2_b, X2,
                                                   (float*)d_out, nullptr,
                                                   1024, 256, 0);
}
